// MultiLayerSSM_18597208391917
// MI455X (gfx1250) — matrix operand order repaired and checked by reading, not compiled, not run on hardware
//
#include <hip/hip_runtime.h>
#include <math.h>

#define HID   512
#define SEQ   784
#define BATCH 1024
#define OUTD  10

#define NB_B 8          // batch blocks (128 rows each)
#define NB_N 8          // hidden blocks (64 cols each)
#define NBLOCKS (NB_B*NB_N)
#define BTILE 128
#define NTILE 64
#define PITCH 520       // bf16 halves per LDS Ad row (512 + 8 pad -> conflict-free b128)

// discretization constant c = step/2 = 1/(2*SEQ)
#define CDISC (1.0f/(2.0f*784.0f))

typedef __attribute__((ext_vector_type(16))) __bf16 v16bf;
typedef __attribute__((ext_vector_type(8)))  __bf16 v8bf;
typedef __attribute__((ext_vector_type(8)))  float  v8f;

__device__ __forceinline__ float Pn(int i){ return sqrtf(1.0f + 2.0f*(float)i); }

__global__ void memset32_kernel(unsigned int* p, int n){
  for (int i = blockIdx.x*blockDim.x + threadIdx.x; i < n; i += gridDim.x*blockDim.x)
    p[i] = 0u;
}

// (I - cA) is lower triangular (HiPPO-LegS A is lower triangular).
// Solve (I - cA) X = (I + cA) column by column -> Ad; column j==HID solves Bd = (I-cA)^{-1} (step*B).
__global__ void hippo_solve_kernel(__bf16* __restrict__ Adbf, float* __restrict__ Bd){
  const int j   = blockIdx.x;
  const int tid = threadIdx.x;
  const bool isB = (j == HID);
  __shared__ float r[HID];

  for (int i = tid; i < HID; i += blockDim.x){
    float v;
    if (isB)         v = (1.0f/784.0f) * Pn(i);              // step * B_i
    else if (i <  j) v = 0.0f;
    else if (i == j) v = 1.0f - CDISC*(1.0f + (float)i);     // 1 + c*A_jj, A_jj = -(1+j)
    else             v = -CDISC * Pn(i) * Pn(j);             // c*A_ij, A_ij = -P_i P_j (i>j)
    r[i] = v;
  }
  __syncthreads();

  const int start = isB ? 0 : j;
  for (int i = start; i < HID; ++i){
    const float Mii = 1.0f + CDISC*(1.0f + (float)i);        // (I - cA)_ii
    const float xi  = r[i] / Mii;                             // uniform across threads
    const float cpi = CDISC * Pn(i);                          // (I - cA)_ki = c*P_k*P_i, k>i
    for (int k = i + 1 + tid; k < HID; k += blockDim.x)
      r[k] -= xi * (cpi * Pn(k));
    if (tid == 0){
      if (isB) Bd[i] = xi;
      else     Adbf[(size_t)i*HID + j] = (__bf16)xi;          // Ad[n=i][k=j], row-major [n][k]
    }
    __syncthreads();
  }
}

// G[n][o] = sum_k C1[k][n] * W[k][o]   (C1^T @ W, 512x10)
__global__ void gemmG_kernel(const float* __restrict__ C1, const float* __restrict__ W,
                             float* __restrict__ G){
  int t = blockIdx.x*blockDim.x + threadIdx.x;
  if (t >= HID*OUTD) return;
  int n = t / OUTD, o = t % OUTD;
  float acc = 0.0f;
  for (int k = 0; k < HID; ++k) acc += C1[(size_t)k*HID + n] * W[k*OUTD + o];
  G[t] = acc;
}

// logits[b][o] = bias[o] + sum_n h_last[b][n] * G[n][o]
__global__ void head_kernel(const __bf16* __restrict__ Hfin, const float* __restrict__ G,
                            const float* __restrict__ bias, float* __restrict__ out){
  int t = blockIdx.x*blockDim.x + threadIdx.x;
  if (t >= BATCH*OUTD) return;
  int b = t / OUTD, o = t % OUTD;
  float acc = bias[o];
  for (int n = 0; n < HID; ++n) acc += (float)Hfin[(size_t)b*HID + n] * G[n*OUTD + o];
  out[t] = acc;
}

// Persistent recurrent layer: H' = tanh(H @ Ad^T + u_t (x) Bd); optionally y0[b,t] = H'.C0
__global__ __launch_bounds__(256, 1)
void ssm_layer_kernel(const __bf16* __restrict__ Adbf,
                      const float*  __restrict__ Bd,
                      const float*  __restrict__ C0,
                      const float*  __restrict__ usrc,   // [BATCH][SEQ]
                      float*        __restrict__ y0out,  // layer0 only (pre-zeroed)
                      __bf16*       __restrict__ Hbuf0,
                      __bf16*       __restrict__ Hbuf1,
                      unsigned int* __restrict__ barCnt,
                      unsigned int* __restrict__ barEpoch,
                      int layer)
{
  extern __shared__ char smem[];
  __bf16* ldsA = (__bf16*)smem;                          // NTILE rows x PITCH halves
  float*  u_l  = (float*)(smem + (size_t)NTILE*PITCH*2); // BTILE
  float*  Bd_l = u_l + BTILE;                            // NTILE
  float*  C0_l = Bd_l + NTILE;                           // NTILE

  const int tid  = threadIdx.x;
  const int lane = tid & 31;
  const int wid  = tid >> 5;
  const int nb   = blockIdx.x % NB_N;
  const int bb   = blockIdx.x / NB_N;
  const int n0   = nb * NTILE;
  const int b0   = bb * BTILE;

  // cache this block's Ad row slice (constant across all 784 steps) in LDS
  for (int idx = tid; idx < NTILE*HID; idx += blockDim.x){
    int nl = idx >> 9;            // /512
    int k  = idx & (HID-1);
    ldsA[nl*PITCH + k] = Adbf[(size_t)(n0+nl)*HID + k];
  }
  for (int i = tid; i < NTILE; i += blockDim.x){
    Bd_l[i] = Bd[n0+i];
    C0_l[i] = C0[n0+i];
  }
  __syncthreads();

  // 8 waves -> 4(m) x 2(n); each wave owns 32 rows x 32 cols = 2x2 WMMA frags
  const int mg = (wid >> 1) * 32;
  const int ng = (wid & 1) * 32;
  const int lm = lane & 15;
  const int lh = lane >> 4;

  for (int t = 0; t < SEQ; ++t){
    if (tid < BTILE) u_l[tid] = usrc[(size_t)(b0+tid)*SEQ + t];
    __syncthreads();

    const __bf16* Hr = (t & 1) ? Hbuf1 : Hbuf0;   // read state
    __bf16*       Hw = (t & 1) ? Hbuf0 : Hbuf1;   // write next state

    v8f acc[2][2] = { { {}, {} }, { {}, {} } };

    #pragma unroll 4
    for (int kt = 0; kt < HID/32; ++kt){
      v16bf aF[2], bF[2];
      const int kA = kt*32 + lh*8;                 // A frag: two 8-half runs (k, k+16)
      #pragma unroll
      for (int i = 0; i < 2; ++i){
        const size_t row = (size_t)(b0 + mg + i*16 + lm) * HID;
        v8bf lo = *(const v8bf*)(Hr + row + kA);
        v8bf hi = *(const v8bf*)(Hr + row + kA + 16);
        aF[i] = __builtin_shufflevector(lo, hi, 0,1,2,3,4,5,6,7,8,9,10,11,12,13,14,15);
      }
      const int kB = kt*32 + lh*8;                 // B frag: two 8-half runs (k, k+16)
      #pragma unroll
      for (int j = 0; j < 2; ++j){
        const int nl = ng + j*16 + lm;
        v8bf lo = *(const v8bf*)(ldsA + nl*PITCH + kB);
        v8bf hi = *(const v8bf*)(ldsA + nl*PITCH + kB + 16);
        bF[j] = __builtin_shufflevector(lo, hi, 0,1,2,3,4,5,6,7,8,9,10,11,12,13,14,15);
      }
      #pragma unroll
      for (int i = 0; i < 2; ++i)
        #pragma unroll
        for (int j = 0; j < 2; ++j)
          acc[i][j] = __builtin_amdgcn_wmma_f32_16x16x32_bf16(
              false, aF[i], false, bF[j], (short)0, acc[i][j], false, false);
    }

    // epilogue: + u*Bd, tanh, store bf16 next-state, C0 partial dot (layer 0)
    #pragma unroll
    for (int i = 0; i < 2; ++i){
      float part[8];
      #pragma unroll
      for (int v = 0; v < 8; ++v) part[v] = 0.0f;
      #pragma unroll
      for (int j = 0; j < 2; ++j){
        const int   nl  = ng + j*16 + lm;
        const float bdn = Bd_l[nl];
        const float c0n = C0_l[nl];
        #pragma unroll
        for (int v = 0; v < 8; ++v){
          const int ml  = mg + i*16 + lh*8 + v;    // C layout: VGPR v -> M = v / v+8
          float val = acc[i][j][v] + u_l[ml] * bdn;
          float h   = tanhf(val);
          Hw[(size_t)(b0+ml)*HID + (n0+nl)] = (__bf16)h;
          part[v] += c0n * h;
        }
      }
      if (layer == 0){
        #pragma unroll
        for (int v = 0; v < 8; ++v){
          float p = part[v];
          p += __shfl_xor(p, 1);
          p += __shfl_xor(p, 2);
          p += __shfl_xor(p, 4);
          p += __shfl_xor(p, 8);                   // reduce 16-lane half (same m)
          if (lm == 0){
            const int ml = mg + i*16 + lh*8 + v;
            atomicAdd(&y0out[(size_t)(b0+ml)*SEQ + t], p);
          }
        }
      }
    }

    // device-wide step barrier (monotonic epoch; 64 co-resident blocks)
    __threadfence();
    __syncthreads();
    if (tid == 0){
      unsigned prev = atomicAdd(barCnt, 1u);
      if (prev == (unsigned)(NBLOCKS - 1)){
        __hip_atomic_store(barCnt, 0u, __ATOMIC_RELAXED, __HIP_MEMORY_SCOPE_AGENT);
        __hip_atomic_fetch_add(barEpoch, 1u, __ATOMIC_RELEASE, __HIP_MEMORY_SCOPE_AGENT);
      } else {
        while (__hip_atomic_load(barEpoch, __ATOMIC_ACQUIRE, __HIP_MEMORY_SCOPE_AGENT)
               < (unsigned)(t+1))
          __builtin_amdgcn_s_sleep(1);
      }
    }
    __syncthreads();
  }
}

extern "C" void kernel_launch(void* const* d_in, const int* in_sizes, int n_in,
                              void* d_out, int out_size, void* d_ws, size_t ws_size,
                              hipStream_t stream){
  (void)in_sizes; (void)n_in; (void)out_size; (void)ws_size;
  const float* x    = (const float*)d_in[0];   // (1024, 784)
  const float* C0   = (const float*)d_in[1];   // (1, 512)
  const float* C1   = (const float*)d_in[2];   // (512, 512)
  const float* W    = (const float*)d_in[3];   // (512, 10)
  const float* bias = (const float*)d_in[4];   // (10,)
  float* out = (float*)d_out;                  // (1024, 10)

  char* ws = (char*)d_ws;
  size_t off = 0;
  auto take = [&](size_t bytes){ size_t r = off; off += (bytes + 255) & ~(size_t)255; return r; };
  __bf16*   Adbf = (__bf16*)(ws + take((size_t)HID*HID*2));   // Ad, bf16, [n][k]
  float*    Bd   = (float*)(ws + take((size_t)HID*4));
  float*    G    = (float*)(ws + take((size_t)HID*OUTD*4));   // C1^T @ W
  float*    y0   = (float*)(ws + take((size_t)BATCH*SEQ*4));  // layer-0 output stream
  __bf16*   H0   = (__bf16*)(ws + take((size_t)BATCH*HID*2)); // state double-buffer
  __bf16*   H1   = (__bf16*)(ws + take((size_t)BATCH*HID*2));
  unsigned* bar  = (unsigned*)(ws + take(256));               // [cnt, epoch]

  const int shmem = NTILE*PITCH*2 + BTILE*4 + NTILE*4 + NTILE*4;  // ~67.6 KB (of 320 KB WGP LDS)

  // init: zero y0 accumulator, initial state h0, barrier words
  memset32_kernel<<<256, 256, 0, stream>>>((unsigned*)y0, BATCH*SEQ);
  memset32_kernel<<<64,  256, 0, stream>>>((unsigned*)H0, BATCH*HID/2);
  memset32_kernel<<<1,    64, 0, stream>>>(bar, 64);

  // build Ad (bf16) and Bd via triangular solves; G = C1^T W
  hippo_solve_kernel<<<HID+1, 256, 0, stream>>>(Adbf, Bd);
  gemmG_kernel<<<(HID*OUTD+255)/256, 256, 0, stream>>>(C1, W, G);

  // layer 0: driven by x, emits y0
  ssm_layer_kernel<<<NBLOCKS, 256, shmem, stream>>>(Adbf, Bd, C0, x, y0, H0, H1, bar, bar+1, 0);

  // reset state + barrier, layer 1: driven by y0 (only final state needed)
  memset32_kernel<<<64, 256, 0, stream>>>((unsigned*)H0, BATCH*HID/2);
  memset32_kernel<<<1,   64, 0, stream>>>(bar, 64);
  ssm_layer_kernel<<<NBLOCKS, 256, shmem, stream>>>(Adbf, Bd, C0, y0, nullptr, H0, H1, bar, bar+1, 1);

  // SEQ=784 even -> final state lands in H0
  head_kernel<<<(BATCH*OUTD+255)/256, 256, 0, stream>>>(H0, G, bias, out);
}
